// AttentionEvaluatorModel_39943195853113
// MI455X (gfx1250) — compile-verified, matrix operand order checked by reading, not run on hardware
//
#include <hip/hip_runtime.h>

// ---------------------------------------------------------------------------
// AttentionEvaluatorModel on MI455X (gfx1250, wave32, WMMA).
// All GEMMs: bf16 x bf16 -> f32 accum via v_wmma_f32_16x16x32_bf16, with
// double-buffered LDS tiles filled by GLOBAL_LOAD_ASYNC_TO_LDS_B128
// (ASYNCcnt-tracked, VGPR-bypassing) overlapping the WMMA stream.
// Pipeline (all on `stream`):
//   cvt:   query/features/weights f32->bf16 ; values f32->bf16 transposed (B,E,F)
//   gate:  gate[b,f] = ftw[b,f] * mask[b,f]
//   G1: fm1 = features @ Wf1^T + bf1          (65536x1024x512)   -> bf16
//   G2: fm  = fm1 @ Wf2^T + bf2               (65536x512x1024)   -> bf16
//   G3: qm1 = query @ Wq1^T + bq1             (1024x1024x512)    -> bf16
//   G4: qm  = qm1 @ Wq2^T + bq2               (1024x512x1024)    -> bf16
//   G5: wts_b = sigmoid(qm_b @ fm_b^T)*gate_b (16 x 64x4096x512) -> bf16
//   G6: pooled_b = wts_b @ valuesT_b^T        (16 x 64x512x4096) -> bf16
//   G7: h   = relu(pooled @ Wc1^T + bc1)      (1024x1024x512)    -> bf16
//   G8: out = h @ Wc2^T + bc2                 (1024x128x1024)    -> f32 d_out
// ---------------------------------------------------------------------------

typedef __bf16 bf16;
typedef __attribute__((ext_vector_type(16))) __bf16 bf16x16;
typedef __attribute__((ext_vector_type(8)))  float  f32x8;

__device__ __forceinline__ bf16 to_bf16(float f) {
  union { float f; unsigned u; } x; x.f = f;
  unsigned r = x.u + 0x7FFFu + ((x.u >> 16) & 1u);   // round-to-nearest-even
  unsigned short h = (unsigned short)(r >> 16);
  union { unsigned short s; bf16 b; } y; y.s = h;
  return y.b;
}

// Async 16B global->LDS copy (VGPR-bypassing, ASYNCcnt-tracked).
// vdst operand = per-lane LDS byte offset; vaddr = per-lane 64-bit address.
__device__ __forceinline__ void async_copy_b128(unsigned lds_off, const void* gptr) {
  asm volatile("global_load_async_to_lds_b128 %0, %1, off"
               :: "v"(lds_off), "v"(gptr)
               : "memory");
}
__device__ __forceinline__ void wait_async0() {
  asm volatile("s_wait_asynccnt 0" ::: "memory");
}
// Low 32 bits of a generic pointer to __shared__ = AS(3) byte offset = LDS addr.
__device__ __forceinline__ unsigned lds_off(const void* p) {
  return (unsigned)(unsigned long long)p;
}

// ---------------------------- elementwise kernels --------------------------

__global__ void cvt_f32_bf16(const float* __restrict__ s, bf16* __restrict__ d,
                             long long n) {
  long long i = (long long)blockIdx.x * blockDim.x + threadIdx.x;
  long long stride = (long long)gridDim.x * blockDim.x;
  for (; i < n; i += stride) d[i] = to_bf16(s[i]);
}

// values (B,F,E) f32 -> valuesT (B,E,F) bf16
__global__ void transpose_cvt(const float* __restrict__ s, bf16* __restrict__ d,
                              int Bn, int Fn, int En) {
  long long n = (long long)Bn * Fn * En;
  long long i = (long long)blockIdx.x * blockDim.x + threadIdx.x;
  long long stride = (long long)gridDim.x * blockDim.x;
  for (; i < n; i += stride) {
    long long b = i / ((long long)Fn * En);
    long long rem = i - b * (long long)Fn * En;
    int f = (int)(rem / En);
    int e = (int)(rem - (long long)f * En);
    d[(b * En + e) * Fn + f] = to_bf16(s[i]);
  }
}

__global__ void gate_kernel(const float* __restrict__ ftw,
                            const unsigned char* __restrict__ mask,
                            float* __restrict__ gate, int n) {
  int i = blockIdx.x * blockDim.x + threadIdx.x;
  if (i < n) gate[i] = mask[i] ? ftw[i] : 0.0f;
}

// ------------------------------- WMMA GEMM ---------------------------------
// C[bz] = epi( A[bz] (MxK bf16 row-major) @ Bw[bz]^T (NxK bf16 row-major)
//              + bias[n] ) ; epi: optional sigmoid*colscale[n], optional relu.
// Requires: M%BM==0, N%BN==0, K%BK==0 (all shapes here satisfy this).

template<int BM, int BN, int BK, int WM, int WN, bool SIG, bool RELU, bool OUTF32>
__global__ __launch_bounds__(256) void gemm_nt(
    const bf16* __restrict__ A, const bf16* __restrict__ Bw,
    const float* __restrict__ bias, const float* __restrict__ colscale,
    void* __restrict__ Cout, int M, int N, int K,
    long long sA, long long sB, long long sC, long long sScale)
{
  constexpr int BLOCK = 256;
  constexpr int TM = BM / WM / 16;     // 16x16 tiles per wave (M)
  constexpr int TN = BN / WN / 16;     // 16x16 tiles per wave (N)
  constexpr int CHK = BK / 8;          // 16B chunks per tile row

  __shared__ bf16 As[2 * BM * BK];     // double-buffered
  __shared__ bf16 Bs[2 * BN * BK];

  const int tid  = threadIdx.x;
  const int lane = tid & 31;
  const int wave = tid >> 5;
  const int wm   = wave % WM;
  const int wn   = wave / WM;
  const int r    = lane & 15;          // N column within 16x16 tile
  const int h8   = (lane >> 4) * 8;    // K-chunk select / row offset per ISA layout

  const int bm0 = blockIdx.y * BM;
  const int bn0 = blockIdx.x * BN;
  const int bz  = blockIdx.z;

  const bf16* Ab = A  + (long long)bz * sA;
  const bf16* Bb = Bw + (long long)bz * sB;
  const float* scb = colscale ? (colscale + (long long)bz * sScale) : nullptr;

  // Issue async copies for K-tile kt into LDS buffer `buf`.
  auto stage = [&](int kt, int buf) {
    const int kb = kt * BK;
    bf16* Asb = As + buf * (BM * BK);
    bf16* Bsb = Bs + buf * (BN * BK);
#pragma unroll
    for (int i = tid; i < BM * CHK; i += BLOCK) {
      int rr = i / CHK, cc = i - rr * CHK;
      async_copy_b128(lds_off(&Asb[rr * BK + cc * 8]),
                      Ab + (long long)(bm0 + rr) * K + kb + cc * 8);
    }
#pragma unroll
    for (int i = tid; i < BN * CHK; i += BLOCK) {
      int rr = i / CHK, cc = i - rr * CHK;
      async_copy_b128(lds_off(&Bsb[rr * BK + cc * 8]),
                      Bb + (long long)(bn0 + rr) * K + kb + cc * 8);
    }
  };

  f32x8 acc[TM][TN];
#pragma unroll
  for (int i = 0; i < TM; ++i)
#pragma unroll
    for (int j = 0; j < TN; ++j)
      acc[i][j] = f32x8{0.f, 0.f, 0.f, 0.f, 0.f, 0.f, 0.f, 0.f};

  const int nk = K / BK;
  stage(0, 0);
  for (int kt = 0; kt < nk; ++kt) {
    const int cur = kt & 1;
    wait_async0();        // my copies into buf[cur] (and earlier) done
    __syncthreads();      // everyone's copies done; prev compute done everywhere
    if (kt + 1 < nk) stage(kt + 1, cur ^ 1);   // overlaps with WMMAs below

    const bf16* Asb = As + cur * (BM * BK);
    const bf16* Bsb = Bs + cur * (BN * BK);
#pragma unroll
    for (int kk = 0; kk < BK; kk += 32) {
      union Frag { uint4 u[2]; bf16x16 v; };
      bf16x16 af[TM], bfm[TN];
#pragma unroll
      for (int i = 0; i < TM; ++i) {
        // ISA 16-bit A layout: lane r row, K chunks {h8..h8+7, 16+h8..16+h8+7}
        const bf16* p = &Asb[(wm * (TM * 16) + i * 16 + r) * BK + kk + h8];
        Frag fr; fr.u[0] = *(const uint4*)p; fr.u[1] = *(const uint4*)(p + 16);
        af[i] = fr.v;
      }
#pragma unroll
      for (int j = 0; j < TN; ++j) {
        const bf16* p = &Bsb[(wn * (TN * 16) + j * 16 + r) * BK + kk + h8];
        Frag fr; fr.u[0] = *(const uint4*)p; fr.u[1] = *(const uint4*)(p + 16);
        bfm[j] = fr.v;
      }
#pragma unroll
      for (int i = 0; i < TM; ++i)
#pragma unroll
        for (int j = 0; j < TN; ++j)
          acc[i][j] = __builtin_amdgcn_wmma_f32_16x16x32_bf16(
              false, af[i], false, bfm[j], (short)0, acc[i][j], false, false);
    }
  }

  // Epilogue: C/D layout -> element (vgpr v, halflane) = row v + 8*(lane>>4), col lane&15
#pragma unroll
  for (int j = 0; j < TN; ++j) {
    const int n  = bn0 + wn * (TN * 16) + j * 16 + r;
    const float bv = bias ? bias[n] : 0.0f;
    const float sv = scb ? scb[n] : 1.0f;
#pragma unroll
    for (int i = 0; i < TM; ++i) {
      const int mbase = bm0 + wm * (TM * 16) + i * 16 + h8;
      f32x8 a = acc[i][j];
#pragma unroll
      for (int v = 0; v < 8; ++v) {
        float x = a[v] + bv;
        if (SIG)  { x = 1.0f / (1.0f + __expf(-x)); x *= sv; }
        if (RELU) x = fmaxf(x, 0.0f);
        long long off = (long long)bz * sC + (long long)(mbase + v) * N + n;
        if (OUTF32) ((float*)Cout)[off] = x;
        else        ((bf16*)Cout)[off]  = to_bf16(x);
      }
    }
  }
}

// ------------------------------- launcher ----------------------------------

extern "C" void kernel_launch(void* const* d_in, const int* in_sizes, int n_in,
                              void* d_out, int out_size, void* d_ws, size_t ws_size,
                              hipStream_t stream) {
  (void)in_sizes; (void)n_in; (void)out_size; (void)ws_size;
  constexpr int Bn = 16, Qn = 64, Fn = 4096, En = 512;
  constexpr int QSn = 512, FSn = 512, Hn = 1024, CHn = 1024, Ln = 128;

  const float* query = (const float*)d_in[0];
  const float* feats = (const float*)d_in[1];
  const float* vals  = (const float*)d_in[2];
  const unsigned char* mask = (const unsigned char*)d_in[3];
  const float* ftw   = (const float*)d_in[4];
  const float* Wq1 = (const float*)d_in[5];  const float* bq1 = (const float*)d_in[6];
  const float* Wq2 = (const float*)d_in[7];  const float* bq2 = (const float*)d_in[8];
  const float* Wf1 = (const float*)d_in[9];  const float* bf1 = (const float*)d_in[10];
  const float* Wf2 = (const float*)d_in[11]; const float* bf2 = (const float*)d_in[12];
  const float* Wc1 = (const float*)d_in[13]; const float* bc1 = (const float*)d_in[14];
  const float* Wc2 = (const float*)d_in[15]; const float* bc2 = (const float*)d_in[16];

  size_t off = 0;
  auto alloc = [&](size_t bytes) -> void* {
    off = (off + 255) & ~(size_t)255;
    void* p = (char*)d_ws + off;
    off += bytes;
    return p;
  };

  bf16* qry_b = (bf16*)alloc((size_t)Bn * Qn * QSn * 2);
  bf16* fea_b = (bf16*)alloc((size_t)Bn * Fn * FSn * 2);
  bf16* valT  = (bf16*)alloc((size_t)Bn * En * Fn * 2);
  bf16* Wq1b  = (bf16*)alloc((size_t)Hn * QSn * 2);
  bf16* Wq2b  = (bf16*)alloc((size_t)En * Hn * 2);
  bf16* Wf1b  = (bf16*)alloc((size_t)Hn * FSn * 2);
  bf16* Wf2b  = (bf16*)alloc((size_t)En * Hn * 2);
  bf16* Wc1b  = (bf16*)alloc((size_t)CHn * En * 2);
  bf16* Wc2b  = (bf16*)alloc((size_t)Ln * CHn * 2);
  float* gate = (float*)alloc((size_t)Bn * Fn * 4);
  bf16* fm1   = (bf16*)alloc((size_t)Bn * Fn * Hn * 2);
  bf16* fm    = (bf16*)alloc((size_t)Bn * Fn * En * 2);
  bf16* qm1   = (bf16*)alloc((size_t)Bn * Qn * Hn * 2);
  bf16* qm    = (bf16*)alloc((size_t)Bn * Qn * En * 2);
  bf16* wts   = (bf16*)alloc((size_t)Bn * Qn * Fn * 2);
  bf16* pool  = (bf16*)alloc((size_t)Bn * Qn * En * 2);
  bf16* hbuf  = (bf16*)alloc((size_t)Bn * Qn * CHn * 2);

  auto cvt = [&](const float* s, bf16* d, long long n) {
    int grid = (int)((n + 1023) / 1024);
    cvt_f32_bf16<<<grid, 256, 0, stream>>>(s, d, n);
  };

  cvt(query, qry_b, (long long)Bn * Qn * QSn);
  cvt(feats, fea_b, (long long)Bn * Fn * FSn);
  cvt(Wq1, Wq1b, (long long)Hn * QSn);
  cvt(Wq2, Wq2b, (long long)En * Hn);
  cvt(Wf1, Wf1b, (long long)Hn * FSn);
  cvt(Wf2, Wf2b, (long long)En * Hn);
  cvt(Wc1, Wc1b, (long long)CHn * En);
  cvt(Wc2, Wc2b, (long long)Ln * CHn);

  {
    long long n = (long long)Bn * Fn * En;
    transpose_cvt<<<(int)((n + 1023) / 1024), 256, 0, stream>>>(vals, valT, Bn, Fn, En);
  }
  gate_kernel<<<(Bn * Fn + 255) / 256, 256, 0, stream>>>(ftw, mask, gate, Bn * Fn);

  // G1: fm1 = features @ Wf1^T + bf1
  gemm_nt<128, 128, 64, 4, 2, false, false, false>
      <<<dim3(Hn / 128, (Bn * Fn) / 128, 1), 256, 0, stream>>>(
          fea_b, Wf1b, bf1, nullptr, fm1, Bn * Fn, Hn, FSn, 0, 0, 0, 0);
  // G2: fm = fm1 @ Wf2^T + bf2
  gemm_nt<128, 128, 64, 4, 2, false, false, false>
      <<<dim3(En / 128, (Bn * Fn) / 128, 1), 256, 0, stream>>>(
          fm1, Wf2b, bf2, nullptr, fm, Bn * Fn, En, Hn, 0, 0, 0, 0);
  // G3: qm1 = query @ Wq1^T + bq1
  gemm_nt<128, 128, 64, 4, 2, false, false, false>
      <<<dim3(Hn / 128, (Bn * Qn) / 128, 1), 256, 0, stream>>>(
          qry_b, Wq1b, bq1, nullptr, qm1, Bn * Qn, Hn, QSn, 0, 0, 0, 0);
  // G4: qm = qm1 @ Wq2^T + bq2
  gemm_nt<128, 128, 64, 4, 2, false, false, false>
      <<<dim3(En / 128, (Bn * Qn) / 128, 1), 256, 0, stream>>>(
          qm1, Wq2b, bq2, nullptr, qm, Bn * Qn, En, Hn, 0, 0, 0, 0);
  // G5: wts_b = sigmoid(qm_b @ fm_b^T) * gate_b   (batched)
  gemm_nt<64, 128, 64, 2, 4, true, false, false>
      <<<dim3(Fn / 128, Qn / 64, Bn), 256, 0, stream>>>(
          qm, fm, nullptr, gate, wts, Qn, Fn, En,
          (long long)Qn * En, (long long)Fn * En, (long long)Qn * Fn, (long long)Fn);
  // G6: pooled_b = wts_b @ valuesT_b^T            (batched)
  gemm_nt<64, 128, 64, 2, 4, false, false, false>
      <<<dim3(En / 128, Qn / 64, Bn), 256, 0, stream>>>(
          wts, valT, nullptr, nullptr, pool, Qn, En, Fn,
          (long long)Qn * Fn, (long long)En * Fn, (long long)Qn * En, 0);
  // G7: h = relu(pooled @ Wc1^T + bc1)
  gemm_nt<128, 128, 64, 4, 2, false, true, false>
      <<<dim3(CHn / 128, (Bn * Qn) / 128, 1), 256, 0, stream>>>(
          pool, Wc1b, bc1, nullptr, hbuf, Bn * Qn, CHn, En, 0, 0, 0, 0);
  // G8: out = h @ Wc2^T + bc2  -> f32 d_out
  gemm_nt<128, 128, 64, 4, 2, false, false, true>
      <<<dim3(Ln / 128, (Bn * Qn) / 128, 1), 256, 0, stream>>>(
          hbuf, Wc2b, bc2, nullptr, d_out, Bn * Qn, Ln, CHn, 0, 0, 0, 0);
}
